// DprnnBlock_stateful_16862041604527
// MI455X (gfx1250) — matrix operand order repaired and checked by reading, compiled, not run on hardware
//
#include <hip/hip_runtime.h>
#include <hip/hip_bf16.h>

typedef __attribute__((ext_vector_type(16))) _Float16 v16h;
typedef __attribute__((ext_vector_type(8)))  _Float16 v8h;
typedef __attribute__((ext_vector_type(4)))  _Float16 v4h;
typedef __attribute__((ext_vector_type(8)))  float    v8f;

static __device__ __forceinline__ float hsig(float x) {
    return fminf(fmaxf(0.2f * x + 0.5f, 0.0f), 1.0f);
}

// CDNA5 has a hardware V_TANH_F32 transcendental; use it if the builtin exists.
static __device__ __forceinline__ float fast_tanh(float x) {
#if __has_builtin(__builtin_amdgcn_tanhf)
    return __builtin_amdgcn_tanhf(x);
#elif __has_builtin(__builtin_amdgcn_tanh_f32)
    return __builtin_amdgcn_tanh_f32(x);
#else
    return tanhf(x);
#endif
}

static __device__ __forceinline__ v8f wmma_f16(const v16h& a, const v16h& b, const v8f& c) {
    return __builtin_amdgcn_wmma_f32_16x16x32_f16(false, a, false, b, (short)0, c, false, false);
}

// ---------------------------------------------------------------------------
// Elementwise helpers
// ---------------------------------------------------------------------------
__global__ void k_f32_to_f16(const float* __restrict__ in, _Float16* __restrict__ out, int n) {
    int i = blockIdx.x * 256 + threadIdx.x;
    if (i < n) out[i] = (_Float16)in[i];
}

// out[r][0..cols-1] = a[r][..], out[r][cols..2cols-1] = b[r][..]   (f32 -> f16)
__global__ void k_pack2_f16(const float* __restrict__ a, const float* __restrict__ b,
                            _Float16* __restrict__ out, int rows, int cols) {
    int i = blockIdx.x * 256 + threadIdx.x;
    int tot = rows * 2 * cols;
    if (i < tot) {
        int r = i / (2 * cols), c = i % (2 * cols);
        out[i] = (_Float16)(c < cols ? a[r * cols + c] : b[r * cols + (c - cols)]);
    }
}

__global__ void k_concat2_f32(const float* __restrict__ a, const float* __restrict__ b,
                              float* __restrict__ out, int n) {
    int i = blockIdx.x * 256 + threadIdx.x;
    if (i < 2 * n) out[i] = (i < n) ? a[i] : b[i - n];
}

// in [L=400, W=128, C=128] f32  ->  out [W=128, L=400, C=128] f16
__global__ void k_transpose_LW(const float* __restrict__ in, _Float16* __restrict__ out) {
    long long o = (long long)blockIdx.x * 256 + threadIdx.x;
    if (o < 400LL * 128 * 128) {
        int c = (int)(o & 127);
        long long row = o >> 7;
        int w = (int)(row / 400);
        int l = (int)(row % 400);
        out[o] = (_Float16)in[((long long)l * 128 + w) * 128 + c];
    }
}

// ---------------------------------------------------------------------------
// Generic tiled WMMA GEMM:  C[M,N] = A[M,K](f16) * B[K,N](f16) + bias[N]
// M % 128 == 0, N % 128 == 0, K % 32 == 0.  256 threads, 128x128 WG tile.
// ---------------------------------------------------------------------------
__global__ void __launch_bounds__(256, 2)
k_gemm_wmma(const _Float16* __restrict__ A, const _Float16* __restrict__ Bm,
            const float* __restrict__ bias, float* __restrict__ C,
            int M, int N, int K) {
    __shared__ _Float16 As[128][40];   // [m][k], padded (row stride 80B, 16B-aligned)
    __shared__ _Float16 Bs[128][40];   // transposed: [n][k]

    int tid = threadIdx.x;
    int wave = tid >> 5, lane = tid & 31;
    int lhi = lane >> 4, llo = lane & 15;
    int bm = blockIdx.x * 128;
    int bn = blockIdx.y * 128;
    int wm = (wave >> 2) * 64;   // 4 M-tiles of 16
    int wn = (wave & 3) * 32;    // 2 N-tiles of 16

    v8f acc[4][2];
    for (int i = 0; i < 4; ++i)
        for (int j = 0; j < 2; ++j) {
            float bv = bias ? bias[bn + wn + j * 16 + llo] : 0.0f;
            for (int v = 0; v < 8; ++v) acc[i][j][v] = bv;
        }

    for (int kb = 0; kb < K; kb += 32) {
        // A tile: 128x32 halves = 512 chunks of 8 halves (16B global + LDS b128)
        for (int i = 0; i < 2; ++i) {
            int cid = tid * 2 + i;                 // 0..511
            int r = cid >> 2, c8 = (cid & 3) * 8;
            *(v8h*)&As[r][c8] = *(const v8h*)&A[(long long)(bm + r) * K + kb + c8];
        }
        // B tile: 32x128 halves; coalesced 16B reads, transpose-scatter to Bs[n][k]
        for (int i = 0; i < 2; ++i) {
            int cid = tid * 2 + i;                 // 0..511
            int k = cid >> 4, n8 = (cid & 15) * 8;
            v8h bv = *(const v8h*)&Bm[(long long)(kb + k) * N + bn + n8];
            for (int u = 0; u < 8; ++u) Bs[n8 + u][k] = bv[u];
        }
        __syncthreads();

        v16h afr[4], bfr[2];
        for (int i = 0; i < 4; ++i) {
            int m = wm + i * 16 + llo;
            for (int j = 0; j < 16; ++j) {
                int kk = (j < 8 ? j : j + 8) + lhi * 8;   // ISA A f16 16x32 layout
                afr[i][j] = As[m][kk];
            }
        }
        for (int j2 = 0; j2 < 2; ++j2) {
            int n = wn + j2 * 16 + llo;
            for (int j = 0; j < 16; ++j) {
                int kk = (j < 8 ? j : j + 8) + lhi * 8;   // ISA B f16 32x16 layout
                bfr[j2][j] = Bs[n][kk];
            }
        }
        for (int i = 0; i < 4; ++i)
            for (int j2 = 0; j2 < 2; ++j2)
                acc[i][j2] = wmma_f16(afr[i], bfr[j2], acc[i][j2]);
        __syncthreads();
    }

    for (int i = 0; i < 4; ++i)
        for (int j2 = 0; j2 < 2; ++j2) {
            int n = bn + wn + j2 * 16 + llo;
            for (int v = 0; v < 8; ++v) {
                int m = bm + wm + i * 16 + v + 8 * lhi;   // ISA D f32 layout
                C[(long long)m * N + n] = acc[i][j2][v];
            }
        }
}

// ---------------------------------------------------------------------------
// Intra-chunk (Bi)LSTM: 16 sequences per WG, hidden 64, 128 time steps.
// xz  [51200, 512]   row = seq*128 + t ; cols [0,256)=fwd gates, [256,512)=bwd
// U   [2][64][256]   f16 (dir-major)
// hout[51200, 128]   f16 ; fwd -> cols [0,64), bwd -> cols [64,128)
// blockIdx.x = seq group (0..24), blockIdx.y = dir (0 fwd, 1 bwd)
// ---------------------------------------------------------------------------
__global__ void __launch_bounds__(256, 1)
k_intra_lstm(const float* __restrict__ xz, const _Float16* __restrict__ U,
             _Float16* __restrict__ hout) {
    __shared__ float    zsh[16][256];
    __shared__ _Float16 hsh[16][64];

    int tid = threadIdx.x, wave = tid >> 5, lane = tid & 31;
    int lhi = lane >> 4, llo = lane & 15;
    int g = blockIdx.x, dir = blockIdx.y;
    const _Float16* Ud = U + dir * 64 * 256;

    // Recurrent weights live in VGPRs for the whole time loop.
    v16h bfr[2][2];                                      // [ntile][ktile]
    for (int nt = 0; nt < 2; ++nt)
        for (int kt = 0; kt < 2; ++kt) {
            int n = wave * 32 + nt * 16 + llo;
            for (int j = 0; j < 16; ++j) {
                int k = kt * 32 + (j < 8 ? j : j + 8) + lhi * 8;
                bfr[nt][kt][j] = Ud[k * 256 + n];
            }
        }

    for (int e = tid; e < 16 * 64; e += 256) (&hsh[0][0])[e] = (_Float16)0.0f;
    float cst[4] = {0.f, 0.f, 0.f, 0.f};
    const int gm = (tid * 4) >> 6;        // this thread's gate-stage row
    const int gch = (tid * 4) & 63;       // this thread's gate-stage col base
    __syncthreads();

    for (int step = 0; step < 128; ++step) {
        int t = dir ? (127 - step) : step;

        // Prefetch next step's pre-activations while this step computes.
        if (step < 127) {
            int tn = dir ? (126 - step) : (step + 1);
            long long prow = (long long)(g * 16 + (tid & 15)) * 128 + tn;
            __builtin_prefetch(&xz[prow * 512 + dir * 256 + (tid >> 4) * 16], 0, 1);
        }

        v16h afr[2];
        for (int kt = 0; kt < 2; ++kt)
            for (int j = 0; j < 16; ++j) {
                int k = kt * 32 + (j < 8 ? j : j + 8) + lhi * 8;
                afr[kt][j] = hsh[llo][k];
            }

        v8f acc[2];
        for (int nt = 0; nt < 2; ++nt) {
            int n = wave * 32 + nt * 16 + llo;
            for (int v = 0; v < 8; ++v) {
                int m = v + 8 * lhi;
                long long row = (long long)(g * 16 + m) * 128 + t;
                acc[nt][v] = xz[row * 512 + dir * 256 + n];
            }
        }
        for (int nt = 0; nt < 2; ++nt) {
            acc[nt] = wmma_f16(afr[0], bfr[nt][0], acc[nt]);
            acc[nt] = wmma_f16(afr[1], bfr[nt][1], acc[nt]);
        }
        for (int nt = 0; nt < 2; ++nt) {
            int n = wave * 32 + nt * 16 + llo;
            for (int v = 0; v < 8; ++v) zsh[v + 8 * lhi][n] = acc[nt][v];
        }
        __syncthreads();

        v4h hvec;
        for (int q = 0; q < 4; ++q) {
            int ch = gch + q;
            float iv = zsh[gm][ch], fv = zsh[gm][64 + ch];
            float gv = zsh[gm][128 + ch], ov = zsh[gm][192 + ch];
            float cc = hsig(fv) * cst[q] + hsig(iv) * fast_tanh(gv);
            cst[q] = cc;
            hvec[q] = (_Float16)(hsig(ov) * fast_tanh(cc));
        }
        *(v4h*)&hsh[gm][gch] = hvec;
        long long row = (long long)(g * 16 + gm) * 128 + t;
        *(v4h*)&hout[row * 128 + dir * 64 + gch] = hvec;
        __syncthreads();
    }
}

// ---------------------------------------------------------------------------
// Inter-chunk LSTM: 16 sequences per WG, hidden 128, 400 time steps.
// xz  [51200, 512]   row = w*400 + t
// U   [128][512]     f16
// hout[51200, 128]   f16 ; row = w*400 + t
// ---------------------------------------------------------------------------
__global__ void __launch_bounds__(256, 1)
k_inter_lstm(const float* __restrict__ xz, const _Float16* __restrict__ U,
             _Float16* __restrict__ hout) {
    __shared__ float    zsh[16][512];
    __shared__ _Float16 hsh[16][128];

    int tid = threadIdx.x, wave = tid >> 5, lane = tid & 31;
    int lhi = lane >> 4, llo = lane & 15;
    int g = blockIdx.x;   // 0..7

    v16h bfr[4][4];                                      // [ntile][ktile], 128 VGPRs
    for (int nt = 0; nt < 4; ++nt)
        for (int kt = 0; kt < 4; ++kt) {
            int n = wave * 64 + nt * 16 + llo;
            for (int j = 0; j < 16; ++j) {
                int k = kt * 32 + (j < 8 ? j : j + 8) + lhi * 8;
                bfr[nt][kt][j] = U[(long long)k * 512 + n];
            }
        }

    for (int e = tid; e < 16 * 128; e += 256) (&hsh[0][0])[e] = (_Float16)0.0f;
    float cst[8];
    for (int q = 0; q < 8; ++q) cst[q] = 0.f;
    const int gm = (tid * 8) >> 7;        // gate-stage row
    const int gch = (tid * 8) & 127;      // gate-stage col base
    __syncthreads();

    for (int t = 0; t < 400; ++t) {
        if (t < 399) {
            long long prow = (long long)(g * 16 + (tid & 15)) * 400 + (t + 1);
            __builtin_prefetch(&xz[prow * 512 + (tid >> 4) * 32], 0, 1);
        }

        v16h afr[4];
        for (int kt = 0; kt < 4; ++kt)
            for (int j = 0; j < 16; ++j) {
                int k = kt * 32 + (j < 8 ? j : j + 8) + lhi * 8;
                afr[kt][j] = hsh[llo][k];
            }

        v8f acc[4];
        for (int nt = 0; nt < 4; ++nt) {
            int n = wave * 64 + nt * 16 + llo;
            for (int v = 0; v < 8; ++v) {
                int m = v + 8 * lhi;
                long long row = (long long)(g * 16 + m) * 400 + t;
                acc[nt][v] = xz[row * 512 + n];
            }
        }
        for (int nt = 0; nt < 4; ++nt)
            for (int kt = 0; kt < 4; ++kt)
                acc[nt] = wmma_f16(afr[kt], bfr[nt][kt], acc[nt]);

        for (int nt = 0; nt < 4; ++nt) {
            int n = wave * 64 + nt * 16 + llo;
            for (int v = 0; v < 8; ++v) zsh[v + 8 * lhi][n] = acc[nt][v];
        }
        __syncthreads();

        v8h hvec;
        for (int q = 0; q < 8; ++q) {
            int ch = gch + q;
            float iv = zsh[gm][ch], fv = zsh[gm][128 + ch];
            float gv = zsh[gm][256 + ch], ov = zsh[gm][384 + ch];
            float cc = hsig(fv) * cst[q] + hsig(iv) * fast_tanh(gv);
            cst[q] = cc;
            hvec[q] = (_Float16)(hsig(ov) * fast_tanh(cc));
        }
        *(v8h*)&hsh[gm][gch] = hvec;
        long long row = (long long)(g * 16 + gm) * 400 + t;
        *(v8h*)&hout[row * 128 + gch] = hvec;
        __syncthreads();
    }
}

// ---------------------------------------------------------------------------
// LayerNorm over [W,C] (16384 elems) per l, plus residual add.
// layoutWL==0 : vin[l*16384 + e]            ([L,W,C])
// layoutWL==1 : vin[((e>>7)*400 + l)*128 + (e&127)]   ([W,L,C])
// out[l*16384 + e] = res[l*16384 + e] + (v - mu) * rsqrt(var+eps) * g[e] + b[e]
// ---------------------------------------------------------------------------
__global__ void k_ln_res(const float* __restrict__ vin, const float* __restrict__ res,
                         const float* __restrict__ gam, const float* __restrict__ bet,
                         float* __restrict__ out, int layoutWL) {
    __shared__ float ssum[256], ssq[256];
    int l = blockIdx.x, tid = threadIdx.x;
    float s = 0.f, sq = 0.f;
    for (int e = tid; e < 16384; e += 256) {
        float v = layoutWL ? vin[((long long)(e >> 7) * 400 + l) * 128 + (e & 127)]
                           : vin[(long long)l * 16384 + e];
        s += v; sq += v * v;
    }
    ssum[tid] = s; ssq[tid] = sq;
    __syncthreads();
    for (int o = 128; o > 0; o >>= 1) {
        if (tid < o) { ssum[tid] += ssum[tid + o]; ssq[tid] += ssq[tid + o]; }
        __syncthreads();
    }
    float mu  = ssum[0] * (1.0f / 16384.0f);
    float var = ssq[0]  * (1.0f / 16384.0f) - mu * mu;
    float rs  = rsqrtf(var + 1e-3f);
    for (int e = tid; e < 16384; e += 256) {
        float v = layoutWL ? vin[((long long)(e >> 7) * 400 + l) * 128 + (e & 127)]
                           : vin[(long long)l * 16384 + e];
        out[(long long)l * 16384 + e] =
            res[(long long)l * 16384 + e] + (v - mu) * rs * gam[e] + bet[e];
    }
}

// ---------------------------------------------------------------------------
// Host-side orchestration
// ---------------------------------------------------------------------------
extern "C" void kernel_launch(void* const* d_in, const int* in_sizes, int n_in,
                              void* d_out, int out_size, void* d_ws, size_t ws_size,
                              hipStream_t stream) {
    (void)in_sizes; (void)n_in; (void)out_size; (void)ws_size;

    const float* x    = (const float*)d_in[0];   // [400,128,128]
    const float* iwf  = (const float*)d_in[1];   // [128,256]
    const float* iuf  = (const float*)d_in[2];   // [64,256]
    const float* ibf  = (const float*)d_in[3];   // [256]
    const float* iwb  = (const float*)d_in[4];
    const float* iub  = (const float*)d_in[5];
    const float* ibb  = (const float*)d_in[6];
    const float* ifcw = (const float*)d_in[7];   // [128,128]
    const float* ifcb = (const float*)d_in[8];   // [128]
    const float* ilng = (const float*)d_in[9];   // [128,128]
    const float* ilnb = (const float*)d_in[10];
    const float* jw   = (const float*)d_in[11];  // [128,512]
    const float* ju   = (const float*)d_in[12];  // [128,512]
    const float* jb   = (const float*)d_in[13];  // [512]
    const float* jfcw = (const float*)d_in[14];  // [128,128]
    const float* jfcb = (const float*)d_in[15];  // [128]
    const float* jlng = (const float*)d_in[16];
    const float* jlnb = (const float*)d_in[17];
    float* out = (float*)d_out;

    char* ws = (char*)d_ws;
    float*    xz    = (float*)   (ws + 0);            // 51200*512*4 = 104857600
    _Float16* x16   = (_Float16*)(ws + 104857600);    // 51200*128*2 = 13107200  (later xj)
    _Float16* h16   = (_Float16*)(ws + 117964800);    // 13107200
    float*    dense = (float*)   (ws + 131072000);    // 26214400
    float*    resid = (float*)   (ws + 157286400);    // 26214400
    _Float16* wbuf  = (_Float16*)(ws + 183500800);    // 262144 halves = 524288 B
    float*    bcomb = (float*)   (ws + 184025088);    // 512 f32

    _Float16* wcomb16 = wbuf;             // [128,512]  = [iwf | iwb]
    _Float16* ucat16  = wbuf + 65536;     // [2][64,256]
    _Float16* ifcw16  = wbuf + 98304;     // [128,128]
    _Float16* jw16    = wbuf + 114688;    // [128,512]
    _Float16* ju16    = wbuf + 180224;    // [128,512]
    _Float16* jfcw16  = wbuf + 245760;    // [128,128]

    const int NX = 400 * 128 * 128;       // 6553600

    // ---- conversions / packing ----
    hipLaunchKernelGGL(k_f32_to_f16, dim3((NX + 255) / 256), dim3(256), 0, stream, x, x16, NX);
    hipLaunchKernelGGL(k_pack2_f16, dim3((128 * 512 + 255) / 256), dim3(256), 0, stream,
                       iwf, iwb, wcomb16, 128, 256);
    hipLaunchKernelGGL(k_concat2_f32, dim3((512 + 255) / 256), dim3(256), 0, stream,
                       ibf, ibb, bcomb, 256);
    hipLaunchKernelGGL(k_f32_to_f16, dim3((64 * 256 + 255) / 256), dim3(256), 0, stream,
                       iuf, ucat16, 64 * 256);
    hipLaunchKernelGGL(k_f32_to_f16, dim3((64 * 256 + 255) / 256), dim3(256), 0, stream,
                       iub, ucat16 + 64 * 256, 64 * 256);
    hipLaunchKernelGGL(k_f32_to_f16, dim3((128 * 128 + 255) / 256), dim3(256), 0, stream,
                       ifcw, ifcw16, 128 * 128);
    hipLaunchKernelGGL(k_f32_to_f16, dim3((128 * 512 + 255) / 256), dim3(256), 0, stream,
                       jw, jw16, 128 * 512);
    hipLaunchKernelGGL(k_f32_to_f16, dim3((128 * 512 + 255) / 256), dim3(256), 0, stream,
                       ju, ju16, 128 * 512);
    hipLaunchKernelGGL(k_f32_to_f16, dim3((128 * 128 + 255) / 256), dim3(256), 0, stream,
                       jfcw, jfcw16, 128 * 128);

    // ---- intra path ----
    // xz = x16 @ [iwf|iwb] + [ibf|ibb]     [51200,512]
    hipLaunchKernelGGL(k_gemm_wmma, dim3(400, 4), dim3(256), 0, stream,
                       x16, wcomb16, bcomb, xz, 51200, 512, 128);
    // bidirectional recurrent pass -> h16 [51200,128] (fwd|bwd concat)
    hipLaunchKernelGGL(k_intra_lstm, dim3(25, 2), dim3(256), 0, stream, xz, ucat16, h16);
    // dense: h16 @ ifcw + ifcb -> dense [51200,128]  ([L,W,C])
    hipLaunchKernelGGL(k_gemm_wmma, dim3(400, 1), dim3(256), 0, stream,
                       h16, ifcw16, ifcb, dense, 51200, 128, 128);
    // LN over [W,C] + residual(x) -> resid
    hipLaunchKernelGGL(k_ln_res, dim3(400), dim3(256), 0, stream,
                       dense, x, ilng, ilnb, resid, 0);

    // ---- inter path ----
    // transpose [L,W,C] -> [W,L,C] f16 (reuse x16 as xj)
    hipLaunchKernelGGL(k_transpose_LW, dim3((NX + 255) / 256), dim3(256), 0, stream, resid, x16);
    // xz = xj @ jw + jb   [51200,512], row = w*400+t
    hipLaunchKernelGGL(k_gemm_wmma, dim3(400, 4), dim3(256), 0, stream,
                       x16, jw16, jb, xz, 51200, 512, 128);
    // recurrent pass over L -> h16 [51200,128], row = w*400+t
    hipLaunchKernelGGL(k_inter_lstm, dim3(8), dim3(256), 0, stream, xz, ju16, h16);
    // dense: h16 @ jfcw + jfcb -> dense [51200,128]  ([W,L,C])
    hipLaunchKernelGGL(k_gemm_wmma, dim3(400, 1), dim3(256), 0, stream,
                       h16, jfcw16, jfcb, dense, 51200, 128, 128);
    // LN over [W,C] (reading [W,L,C] layout) + residual(resid) -> out [L,W,C]
    hipLaunchKernelGGL(k_ln_res, dim3(400), dim3(256), 0, stream,
                       dense, resid, jlng, jlnb, out, 1);
}
